// SimpleLSTModel_14431090114678
// MI455X (gfx1250) — matrix operand order repaired and checked by reading, not compiled, not run on hardware
//
#include <hip/hip_runtime.h>

// LSTM rollout for MI455X (gfx1250), wave32 + WMMA bf16.
// Persistent kernel: each block owns BT=32 batch rows, loops all T steps.
// gates = h @ W_hh^T via V_WMMA_F32_16X16X32_BF16; c in registers (D-layout),
// h ping-pongs in LDS as bf16; W_hh converted once to bf16 (L2-resident).

#define HID 512
#define G4  (4 * HID)        // 2048 gate columns
#define BT  32               // batch rows per block
#define HS  520              // padded LDS row stride (bf16 elems) -> bank spread

typedef __attribute__((ext_vector_type(16))) __bf16 v16bf;
typedef __attribute__((ext_vector_type(8)))  __bf16 v8bf;
typedef __attribute__((ext_vector_type(8)))  float  v8f;

__device__ __forceinline__ unsigned short f2bf(float f) {
  unsigned int u = __float_as_uint(f);
  u += 0x7FFFu + ((u >> 16) & 1u);          // round-to-nearest-even
  return (unsigned short)(u >> 16);
}
__device__ __forceinline__ float bf2f(unsigned short u) {
  return __uint_as_float(((unsigned int)u) << 16);
}
__device__ __forceinline__ float sigm(float x) {
  return 1.0f / (1.0f + __expf(-x));
}
__device__ __forceinline__ float tanh_fast(float x) {
  float c = fminf(fmaxf(x, -15.0f), 15.0f);
  float e = __expf(2.0f * c);
  return (e - 1.0f) / (e + 1.0f);
}

// One-time prep: W_hh f32 -> bf16, and fused bias b_ih + b_hh.
__global__ void lstm_prep(const float* __restrict__ Whh,
                          const float* __restrict__ b_ih,
                          const float* __restrict__ b_hh,
                          unsigned short* __restrict__ Wbf,
                          float* __restrict__ bsum) {
  int i = blockIdx.x * blockDim.x + threadIdx.x;
  if (i < G4 * HID) Wbf[i] = f2bf(Whh[i]);
  if (i < G4)       bsum[i] = b_ih[i] + b_hh[i];
}

__global__ __launch_bounds__(512)
void lstm_roll(const unsigned short* __restrict__ Wbf,   // [2048][512] bf16
               const float* __restrict__ bsum,           // [2048] b_ih+b_hh
               const float* __restrict__ W_ih,           // [2048] (IN=1)
               const float* __restrict__ W_lin,          // [512]  (OUT=1)
               const float* __restrict__ b_lin,          // [1]
               const float* __restrict__ answers,        // [B][T_ans]
               const int*   __restrict__ tf_mask,        // [T_ans]
               const int*   __restrict__ train_sz,       // [1]
               float* __restrict__ out,                  // [B][T_run]
               int T_ans, int T_run) {
  __shared__ __align__(32) unsigned short h_s[2][BT * HS]; // ping-pong h (bf16)
  __shared__ float inp_s[BT];
  __shared__ float partial_s[BT];
  __shared__ float out_s[BT];

  const int tid   = (int)threadIdx.x;
  const int wave  = tid >> 5;          // 0..15
  const int lane  = tid & 31;
  const int l15   = lane & 15;
  const int hiH   = lane >> 4;         // 0 for lanes 0-15, 1 for 16-31
  const int btile = (int)blockIdx.x * BT;
  const int train = train_sz[0];
  const float blin = b_lin[0];

  // init: h0 = 0, out0 = 0, c0 = 0
  for (int i = tid; i < BT * HS; i += 512) h_s[0][i] = 0;
  if (tid < BT) out_s[tid] = 0.0f;
  float c_r[2][2][8];
#pragma unroll
  for (int a = 0; a < 2; ++a)
#pragma unroll
    for (int b = 0; b < 2; ++b)
#pragma unroll
      for (int v = 0; v < 8; ++v) c_r[a][b][v] = 0.0f;

  for (int t = 0; t < T_run; ++t) {
    const int cur = t & 1, nxt = cur ^ 1;
    if (tid < BT) {
      const bool use = (t < train) || (tf_mask[t] > 0);
      inp_s[tid] = use ? answers[(size_t)(btile + tid) * T_ans + t] : out_s[tid];
      partial_s[tid] = 0.0f;
    }
    __syncthreads();   // (A) inp ready; h_s[cur] complete from previous step

    const unsigned short* hcur = &h_s[cur][0];
    unsigned short*       hnxt = &h_s[nxt][0];
    // A-fragment base for this lane (row l15 of M-tile 0; K sub-offset by half)
    const unsigned short* hrowA = hcur + (size_t)l15 * HS + hiH * 8;

#pragma unroll
    for (int ht2 = 0; ht2 < 2; ++ht2) {
      const int ht = wave + 16 * ht2;          // h-column tile 0..31
      const int hc = ht * 16 + l15;            // this lane's h column (N index)

      v8f acc0[4], acc1[4];
#pragma unroll
      for (int g = 0; g < 4; ++g) {
        v8f z = {0.f, 0.f, 0.f, 0.f, 0.f, 0.f, 0.f, 0.f};
        acc0[g] = z; acc1[g] = z;
      }

      const unsigned short* wrow = Wbf + (size_t)hc * HID + hiH * 8;

#pragma unroll 4
      for (int kc = 0; kc < 16; ++kc) {
        // A fragments (16-bit A layout: lanes<16 hold K{0..7,16..23}, lanes>=16 K{8..15,24..31})
        const unsigned short* hb0 = hrowA + kc * 32;
        v8bf lo0 = *(const v8bf*)(hb0);
        v8bf hi0 = *(const v8bf*)(hb0 + 16);
        v16bf a0 = __builtin_shufflevector(lo0, hi0, 0,1,2,3,4,5,6,7,8,9,10,11,12,13,14,15);
        const unsigned short* hb1 = hb0 + 16 * HS;          // M-tile 1 (rows 16..31)
        v8bf lo1 = *(const v8bf*)(hb1);
        v8bf hi1 = *(const v8bf*)(hb1 + 16);
        v16bf a1 = __builtin_shufflevector(lo1, hi1, 0,1,2,3,4,5,6,7,8,9,10,11,12,13,14,15);

        const unsigned short* wk = wrow + kc * 32;
#pragma unroll
        for (int g = 0; g < 4; ++g) {
          // B 32x16: lane = N (gate col), K{0..7,16..23} / K{8..15,24..31} per half, matching A
          const unsigned short* wg4 = wk + (size_t)g * (HID * HID);
          v8bf blo = *(const v8bf*)(wg4);
          v8bf bhi = *(const v8bf*)(wg4 + 16);
          v16bf bfrag = __builtin_shufflevector(blo, bhi, 0,1,2,3,4,5,6,7,8,9,10,11,12,13,14,15);
          acc0[g] = __builtin_amdgcn_wmma_f32_16x16x32_bf16(
              false, a0, false, bfrag, (short)0, acc0[g], false, false);
          acc1[g] = __builtin_amdgcn_wmma_f32_16x16x32_bf16(
              false, a1, false, bfrag, (short)0, acc1[g], false, false);
        }
      }

      // Epilogue: per-lane gate column hc is fixed; rows vary over (mt, v).
      const float bi  = bsum[hc],        bff = bsum[512 + hc];
      const float bgg = bsum[1024 + hc], bo  = bsum[1536 + hc];
      const float wi  = W_ih[hc],        wf  = W_ih[512 + hc];
      const float wg  = W_ih[1024 + hc], wo  = W_ih[1536 + hc];

#pragma unroll
      for (int mt = 0; mt < 2; ++mt) {
#pragma unroll
        for (int v = 0; v < 8; ++v) {
          const int m = mt * 16 + hiH * 8 + v;   // batch row within tile (D layout)
          const float inp = inp_s[m];
          float gi = (mt ? acc1[0][v] : acc0[0][v]) + bi  + inp * wi;
          float gf = (mt ? acc1[1][v] : acc0[1][v]) + bff + inp * wf;
          float gg = (mt ? acc1[2][v] : acc0[2][v]) + bgg + inp * wg;
          float go = (mt ? acc1[3][v] : acc0[3][v]) + bo  + inp * wo;
          float c = sigm(gf) * c_r[ht2][mt][v] + sigm(gi) * tanh_fast(gg);
          c_r[ht2][mt][v] = c;
          float h = sigm(go) * tanh_fast(c);
          hnxt[(size_t)m * HS + hc] = f2bf(h);
        }
      }
    }
    __syncthreads();   // (B) h_s[nxt] complete

    // Output projection: out[m] = sum_k h_new[m,k] * W_lin[k] + b_lin
    {
      const int m  = tid >> 4;          // 0..31
      const int ks = (tid & 15) * 32;   // 0..480
      float s = 0.0f;
#pragma unroll 8
      for (int k = 0; k < 32; ++k)
        s += bf2f(hnxt[(size_t)m * HS + ks + k]) * W_lin[ks + k];
      atomicAdd(&partial_s[m], s);
    }
    __syncthreads();   // (C) partial sums complete

    if (tid < BT) {
      const float o = partial_s[tid] + blin;
      out_s[tid] = o;
      out[(size_t)(btile + tid) * T_run + t] = o;
    }
    // out_s read next iter only by same thread -> no extra barrier needed
  }
}

extern "C" void kernel_launch(void* const* d_in, const int* in_sizes, int n_in,
                              void* d_out, int out_size, void* d_ws, size_t ws_size,
                              hipStream_t stream) {
  (void)n_in; (void)ws_size;
  // d_in order: answers, W_ih, b_ih, W_hh, b_hh, W_lin, b_lin, tf_mask, train_size, future_size
  const float* answers = (const float*)d_in[0];
  const float* W_ih    = (const float*)d_in[1];
  const float* b_ih    = (const float*)d_in[2];
  const float* W_hh    = (const float*)d_in[3];
  const float* b_hh    = (const float*)d_in[4];
  const float* W_lin   = (const float*)d_in[5];
  const float* b_lin   = (const float*)d_in[6];
  const int*   tf_mask = (const int*)d_in[7];
  const int*   trainp  = (const int*)d_in[8];
  float* out = (float*)d_out;

  const int T_ans = in_sizes[7];            // tf_mask has length T
  const int B     = in_sizes[0] / T_ans;    // answers is [B, T]
  const int T_run = out_size / B;           // output is [B, T_run, 1]

  unsigned short* Wbf  = (unsigned short*)d_ws;
  float*          bsum = (float*)((char*)d_ws + (size_t)G4 * HID * sizeof(unsigned short));

  lstm_prep<<<(G4 * HID + 255) / 256, 256, 0, stream>>>(W_hh, b_ih, b_hh, Wbf, bsum);

  lstm_roll<<<B / BT, 512, 0, stream>>>(Wbf, bsum, W_ih, W_lin, b_lin,
                                        answers, tf_mask, trainp, out,
                                        T_ans, T_run);
}
